// I3NN_22574348108021
// MI455X (gfx1250) — matrix operand order repaired and checked by reading, not compiled, not run on hardware
//
#include <hip/hip_runtime.h>
#include <hip/hip_bf16.h>

// ---------------------------------------------------------------------------
// E(3)-equivariant GNN forward for MI455X (gfx1250, wave32, WMMA).
// Heavy GEMMs via v_wmma_f32_16x16x32_f16; FC2+TensorProduct fused so the
// 394MB per-edge weight tensor is never materialized. w2 is pre-transposed to
// f16 [col][k] so B fragments load as global_load_b128 (double-buffered).
// h tiles staged into LDS via the Tensor Data Mover (6-arg builtin).
// ---------------------------------------------------------------------------

typedef __attribute__((ext_vector_type(16))) _Float16 v16h;
typedef __attribute__((ext_vector_type(8)))  _Float16 v8h;
typedef __attribute__((ext_vector_type(8)))  float    v8f;
typedef __attribute__((ext_vector_type(4)))  unsigned int u32x4;
typedef __attribute__((ext_vector_type(8)))  int      i32x8;
typedef __attribute__((ext_vector_type(4)))  int      i32x4;

#if defined(__has_builtin)
#if __has_builtin(__builtin_amdgcn_tensor_load_to_lds) && __has_builtin(__builtin_amdgcn_s_wait_tensorcnt)
#define HAVE_TDM 1
#endif
#endif

#define N_NODES 512
#define N_EDGES 2048
#define NS      128
#define MAXD    448   // max node_attr width across layers
#define CSPLIT  8     // column-range splits per edge block

// ---------------- TP path descriptors (constant, compile-time) -------------
struct TpPath { int woff; int moshift; int xoff; int ooff; int d3; int kind; float pw; };
// kind: 0 scalar x*1 ; 1 x(scalar)*sh_l1 ; 2 x(l1)*sh0 ; 3 dot(x,sh_l1)
//       4 cross(x,sh_l1)/sqrt6 ; 5 l1 x l2 -> l1 (wigner 112^T) path
__constant__ TpPath c_paths[14] = {
  // layer 0 (numel 20480)
  {0,     7, 0,   0,   1, 0, 0.0883883476f},
  {16384, 5, 0,   256, 3, 1, 0.1530931089f},
  // layer 1 (numel 48128)
  {0,     7, 0,   0,   1, 0, 0.0790569415f},
  {16384, 5, 0,   256, 3, 1, 0.1250000000f},
  {20480, 7, 128, 128, 1, 0, 0.0883883476f},
  {36864, 5, 128, 352, 3, 1, 0.1369306394f},
  {40960, 5, 256, 256, 3, 2, 0.1250000000f},
  {41984, 7, 256, 0,   1, 3, 0.0790569415f},
  {46080, 5, 256, 352, 3, 4, 0.1369306394f},
  {47104, 5, 256, 256, 3, 5, 0.1250000000f},
  // layer 2 (numel 40960)
  {0,     7, 0,   0,   1, 0, 0.0790569415f},
  {16384, 7, 128, 128, 1, 0, 0.0790569415f},
  {32768, 7, 256, 0,   1, 3, 0.0790569415f},
  {36864, 7, 352, 128, 1, 3, 0.0790569415f},
};
__constant__ int c_poff[3] = {0, 2, 10};
__constant__ int c_pcnt[3] = {2, 8, 4};

// BatchNorm irrep blocks per layer: {offset, mul, 2l+1, subtract_mean}
struct IrrBlk { int off; int m; int d; int sub; };
__constant__ IrrBlk c_irr[9] = {
  {0,128,1,1},{128,128,1,0},{256,32,3,0},              // L0 -> 352
  {0,128,1,1},{128,128,1,0},{256,32,3,0},{352,32,3,0}, // L1 -> 448
  {0,128,1,1},{128,128,1,0},                            // L2 -> 256
};
__constant__ int c_ioff[3] = {0,3,7};
__constant__ int c_icnt[3] = {3,4,2};

// ---------------- WMMA fragment helpers (contiguous k = 16*(lane>=16) + j in A and B) ---------
__device__ inline v16h frag_a_lds(const _Float16* base, int ld, int lane) {
  // A 16x32 f16: lane holds row M=lane&15; K = 16*(lane>=16) + j
  int m  = lane & 15;
  int kb = (lane >> 4) << 4;
  v16h a;
#pragma unroll
  for (int j = 0; j < 16; ++j)
    a[j] = base[m * ld + kb + j];
  return a;
}
__device__ inline v16h frag_b_lds(const _Float16* base, int ld, int lane) {
  // B 32x16 f16: lane holds col N=lane&15; K = 16*(lane>=16) + j
  int n  = lane & 15;
  int kb = (lane >> 4) << 4;
  v16h b;
#pragma unroll
  for (int j = 0; j < 16; ++j)
    b[j] = base[(kb + j) * ld + n];
  return b;
}

// ---------------- TDM: DMA one 16x384 f16 tile (12KB, contiguous) to LDS ---
__device__ inline void tdm_load_h_tile(const _Float16* gsrc, _Float16* lds_dst) {
#ifdef HAVE_TDM
  unsigned long long ga = (unsigned long long)gsrc;
  unsigned lds_off = (unsigned)(unsigned long long)(void*)lds_dst;
  u32x4 g0;
  g0[0] = 1u;                                   // count=1, user descriptor
  g0[1] = lds_off;                              // lds_addr
  g0[2] = (unsigned)(ga & 0xFFFFFFFFull);       // global_addr[31:0]
  g0[3] = (unsigned)((ga >> 32) & 0x1FFFFFFull) | (2u << 30); // addr[56:32], type=2
  i32x8 g1;
  g1[0] = (1 << 16);     // workgroup_mask=0, data_size=2B
  g1[1] = (6144 << 16);  // tensor_dim0 = 6144 (1-D view of the tile)
  g1[2] = 0;             // tensor_dim0 hi, tensor_dim1 = 0
  g1[3] = (6144 << 16);  // tile_dim0 = 6144
  g1[4] = 0;             // tile_dim1/2 unused
  g1[5] = 6144;          // tensor_dim0_stride
  g1[6] = 0;
  g1[7] = 0;
  i32x4 z4 = {0, 0, 0, 0};
  i32x8 z8 = {0, 0, 0, 0, 0, 0, 0, 0};
  __builtin_amdgcn_tensor_load_to_lds(g0, g1, z4, z4, z8, 0);
  __builtin_amdgcn_s_wait_tensorcnt(0);
#else
  (void)gsrc; (void)lds_dst;
#endif
}

// ---------------- small utility kernels ------------------------------------
__global__ void fill_kernel(float* p, int n, float v) {
  int i = blockIdx.x * blockDim.x + threadIdx.x;
  if (i < n) p[i] = v;
}
__global__ void copy_nodef_kernel(const float* nf, float* na) {
  int i = blockIdx.x * blockDim.x + threadIdx.x;
  if (i >= N_NODES * NS) return;
  na[(i / NS) * MAXD + (i % NS)] = nf[i];
}
__global__ void count_kernel(const int* esrc, float* cnt) {
  int e = blockIdx.x * blockDim.x + threadIdx.x;
  if (e < N_EDGES) atomicAdd(&cnt[esrc[e]], 1.0f);
}

// w2t[col][k] = (f16) w2[k][col] ; 32x32 LDS tile transpose
__global__ __launch_bounds__(256)
void transpose_w2_kernel(const float* __restrict__ w2, _Float16* __restrict__ w2t,
                         int numel) {
  __shared__ float tile[32][33];
  int bc = blockIdx.x * 32;   // column base (numel dim)
  int bk = blockIdx.y * 32;   // k base (384 dim)
  int tid = threadIdx.x;
  int tc = tid & 31, tr = tid >> 5;
  for (int r = tr; r < 32; r += 8)
    tile[r][tc] = w2[(size_t)(bk + r) * numel + (bc + tc)];
  __syncthreads();
  for (int r = tr; r < 32; r += 8)
    w2t[(size_t)(bc + r) * 384 + (bk + tc)] = (_Float16)tile[tc][r];
}

// ---------------- edge geometry + SH + embedding MLP -----------------------
__global__ __launch_bounds__(128)
void edge_init_kernel(const float* edge_f, const float* pos, const float* nuv,
                      const int* esrc, const int* edst,
                      const float* w1, const float* b1,
                      const float* w2, const float* b2,
                      float* edge_sh, float* edge_attr) {
  __shared__ float attr0[33];
  __shared__ float ea[NS];
  int e = blockIdx.x, t = threadIdx.x;
  int s = esrc[e], d = edst[e];
  float px = pos[d*3+0] - pos[s*3+0];
  float py = pos[d*3+1] - pos[s*3+1];
  float pz = pos[d*3+2] - pos[s*3+2];
  float dist = sqrtf(px*px + py*py + pz*pz);
  float inv = 1.0f / fmaxf(dist, 1e-12f);
  float ex = px*inv, ey = py*inv, ez = pz*inv;
  float ca = nuv[d*3+0]*ex + nuv[d*3+1]*ey + nuv[d*3+2]*ez;
  float cb = nuv[s*3+0]*ex + nuv[s*3+1]*ey + nuv[s*3+2]*ez;
  float alpha = acosf(fminf(fmaxf(ca, -1.0f), 1.0f));
  float beta  = acosf(fminf(fmaxf(cb, -1.0f), 1.0f));
  float sa = sinf(alpha), cca = cosf(alpha);
  float sb = sinf(beta),  ccb = cosf(beta);
  float X = sa * sb, Y = ccb, Z = cca * sb;
  if (t < 9) {
    const float s3 = 1.73205080757f, s15 = 3.87298334621f, s5 = 2.2360679775f;
    float v;
    switch (t) {
      case 0: v = 1.0f; break;
      case 1: v = s3 * Y; break;
      case 2: v = s3 * Z; break;
      case 3: v = s3 * X; break;
      case 4: v = s15 * X * Y; break;
      case 5: v = s15 * Y * Z; break;
      case 6: v = 0.5f * s5 * (3.0f * Z * Z - 1.0f); break;
      case 7: v = s15 * Z * X; break;
      default: v = 0.5f * s15 * (X * X - Y * Y); break;
    }
    edge_sh[e * 9 + t] = v;
  }
  if (t == 0) attr0[0] = edge_f[e];
  else if (t < 33) {
    const float step = 30.0f / 31.0f;
    const float coeff = -0.5f / (step * step);
    float off = step * (float)(t - 1);
    float dd = dist - off;
    attr0[t] = __expf(coeff * dd * dd);
  }
  __syncthreads();
  float acc = b1[t];
  for (int i = 0; i < 33; ++i) acc += attr0[i] * w1[i * NS + t];
  ea[t] = fmaxf(acc, 0.0f);
  __syncthreads();
  acc = b2[t];
  for (int i = 0; i < NS; ++i) acc += ea[i] * w2[i * NS + t];
  edge_attr[e * NS + t] = acc;
}

// ---------------- eaug = [edge_attr | na[src,:128] | na[dst,:128]] ---------
__global__ __launch_bounds__(128)
void build_eaug_kernel(const float* eattr, const float* na,
                       const int* esrc, const int* edst, float* eaug) {
  int e = blockIdx.x, t = threadIdx.x;
  int s = esrc[e], d = edst[e];
  eaug[e * 384 + t]         = eattr[e * NS + t];
  eaug[e * 384 + 128 + t]   = na[s * MAXD + t];
  eaug[e * 384 + 256 + t]   = na[d * MAXD + t];
}

// ---------------- generic WMMA GEMM: out = [relu](A @ W + b) ---------------
template<bool RELU, bool OUTH>
__global__ __launch_bounds__(128)
void gemm_kernel(const float* __restrict__ A, int lda,
                 const float* __restrict__ W, const float* __restrict__ bias,
                 void* out, int M, int N, int K) {
  __shared__ _Float16 sA[32][40];
  __shared__ _Float16 sB[32][40];
  int tid = threadIdx.x;
  int bn = blockIdx.x, bm = blockIdx.y;
  int wid = tid >> 5, lane = tid & 31;
  int wr = (wid >> 1) * 16, wc = (wid & 1) * 16;
  v8f c = {};
  for (int k0 = 0; k0 < K; k0 += 32) {
    for (int i = tid; i < 32 * 32; i += 128) {
      int r = i >> 5, cc = i & 31;
      int gm = bm * 32 + r, gk = k0 + cc;
      sA[r][cc] = (_Float16)((gm < M && gk < K) ? A[(size_t)gm * lda + gk] : 0.0f);
      int gk2 = k0 + r, gn = bn * 32 + cc;
      sB[r][cc] = (_Float16)((gk2 < K && gn < N) ? W[(size_t)gk2 * N + gn] : 0.0f);
    }
    __syncthreads();
    v16h a = frag_a_lds(&sA[wr][0], 40, lane);
    v16h b = frag_b_lds(&sB[0][wc], 40, lane);
    c = __builtin_amdgcn_wmma_f32_16x16x32_f16(false, a, false, b, (short)0, c,
                                               false, false);
    __syncthreads();
  }
  int n = bn * 32 + wc + (lane & 15);
  float bv = (n < N) ? bias[n] : 0.0f;
#pragma unroll
  for (int v = 0; v < 8; ++v) {
    int m = bm * 32 + wr + v + ((lane >> 4) << 3);
    if (m < M && n < N) {
      float val = c[v] + bv;
      if (RELU) val = fmaxf(val, 0.0f);
      if (OUTH) ((_Float16*)out)[(size_t)m * N + n] = (_Float16)val;
      else      ((float*)out)[(size_t)m * N + n] = val;
    }
  }
}

// ---------------- on-the-fly Wigner contraction t[e,u,k] -------------------
__device__ inline void eval_t(int kind, int u, int xoff,
                              const _Float16* X, const float* S, float* t) {
  const float R3 = 0.57735026919f;   // 1/sqrt(3)
  const float R6 = 0.40824829046f;   // 1/sqrt(6)
  const float H2 = 0.70710678119f;   // 1/sqrt(2)
  const float B6 = 0.81649658092f;   // 2/sqrt(6)
  const float R5 = 0.44721359550f;   // 1/sqrt(5)
  switch (kind) {
    case 0: t[0] = (float)X[xoff + u]; break;
    case 1: { float x = (float)X[xoff + u];
              t[0] = x * S[1] * R3; t[1] = x * S[2] * R3; t[2] = x * S[3] * R3; } break;
    case 2: { int b = xoff + 3 * u;
              t[0] = (float)X[b] * R3; t[1] = (float)X[b+1] * R3; t[2] = (float)X[b+2] * R3; } break;
    case 3: { int b = xoff + 3 * u;
              t[0] = ((float)X[b]*S[1] + (float)X[b+1]*S[2] + (float)X[b+2]*S[3]) * R3; } break;
    case 4: { int b = xoff + 3 * u;
              float x0 = X[b], x1 = X[b+1], x2 = X[b+2];
              float s0 = S[1], s1 = S[2], s2 = S[3];
              t[0] = (x1 * s2 - x2 * s1) * R6;
              t[1] = (x2 * s0 - x0 * s2) * R6;
              t[2] = (x0 * s1 - x1 * s0) * R6; } break;
    default: { int b = xoff + 3 * u;                        // (1,2,1) path
              float x0 = X[b], x1 = X[b+1], x2 = X[b+2];
              float q0 = S[4], q1 = S[5], q2 = S[6], q3 = S[7], q4 = S[8];
              t[0] = R5 * (H2 * (x2 * q0 + x1 * q1) - R6 * x0 * q2 - H2 * x0 * q4);
              t[1] = R5 * (H2 * (x0 * q1 + x2 * q3) + B6 * x1 * q2);
              t[2] = R5 * (H2 * (x0 * q0 + x1 * q3 + x2 * q4) - R6 * x2 * q2); } break;
  }
}

// ---------------- shared staging + consume logic for FC2+TP ----------------
struct TpShared {
  _Float16 h[16][384];
  _Float16 x[16][MAXD];
  float    sh[16][9];
  float    out[16][MAXD];
};

__device__ inline void tp_stage(TpShared& S, const _Float16* h, const float* na,
                                const float* sh, const int* edst, int eb,
                                int in_dim) {
  int tid = threadIdx.x;
#ifdef HAVE_TDM
  if (tid < 32) tdm_load_h_tile(h + (size_t)eb * 16 * 384, &S.h[0][0]);
#else
  for (int i = tid; i < 16 * 384; i += 256)
    S.h[i / 384][i % 384] = h[(size_t)(eb * 16 + i / 384) * 384 + (i % 384)];
#endif
  for (int i = tid; i < 16 * MAXD; i += 256) {
    int e = i / MAXD, c2 = i % MAXD;
    int gd = edst[eb * 16 + e];
    S.x[e][c2] = (_Float16)((c2 < in_dim) ? na[(size_t)gd * MAXD + c2] : 0.0f);
    S.out[e][c2] = 0.0f;
  }
  for (int i = tid; i < 16 * 9; i += 256)
    S.sh[i / 9][i % 9] = sh[(size_t)(eb * 16 + i / 9) * 9 + (i % 9)];
  __syncthreads();
}

__device__ inline void tp_consume(TpShared& S, const v8f& acc, float bias,
                                  int col, int lane, int layer) {
  const int po = c_poff[layer], pn = c_pcnt[layer];
  TpPath P = c_paths[po];
  for (int i2 = 1; i2 < pn; ++i2)
    if (col >= c_paths[po + i2].woff) P = c_paths[po + i2];
  int q = col - P.woff;
  int u = q >> P.moshift;
  int w = q & ((1 << P.moshift) - 1);
#pragma unroll
  for (int v = 0; v < 8; ++v) {
    int el = v + ((lane >> 4) << 3);          // local edge index 0..15
    float tpw = acc[v] + bias;
    float t[3];
    eval_t(P.kind, u, P.xoff, &S.x[el][0], &S.sh[el][0], t);
    int obase = P.ooff + w * P.d3;
    for (int k = 0; k < P.d3; ++k)
      atomicAdd(&S.out[el][obase + k], P.pw * t[k] * tpw);
  }
}

__device__ inline void tp_scatter(TpShared& S, const int* esrc, float* node_accum,
                                  int eb, int out_dim) {
  __syncthreads();
  int tid = threadIdx.x;
  for (int i = tid; i < 16 * out_dim; i += 256) {
    int e = i / out_dim, d2 = i % out_dim;
    int gs = esrc[eb * 16 + e];
    float v = S.out[e][d2];
    if (v != 0.0f) atomicAdd(&node_accum[(size_t)gs * MAXD + d2], v);
  }
}

// ------ fused FC2 (WMMA) + TP, B from f16-transposed w2t (b128 loads) ------
__global__ __launch_bounds__(256)
void fc2_tp_t_kernel(const _Float16* __restrict__ h, const _Float16* __restrict__ w2t,
                     const float* __restrict__ b2, const float* __restrict__ na,
                     const float* __restrict__ sh,
                     const int* esrc, const int* edst,
                     float* node_accum, int layer, int numel, int in_dim, int out_dim) {
  __shared__ TpShared S;
  int eb = blockIdx.y;
  tp_stage(S, h, na, sh, edst, eb, in_dim);

  int tid = threadIdx.x;
  int wid = tid >> 5, lane = tid & 31;
  v16h afrag[12];
#pragma unroll
  for (int ks = 0; ks < 12; ++ks)
    afrag[ks] = frag_a_lds(&S.h[0][ks * 32], 384, lane);

  int nchunks = numel >> 4;
  int stride  = gridDim.x * 8;
  int n   = lane & 15;
  int bb  = (lane >> 4) << 1;  // kb2/8: 0 or 2, in v8h units

  for (int ch = blockIdx.x * 8 + wid; ch < nchunks; ch += stride) {
    int col = ch * 16 + n;
    const v8h* wp = (const v8h*)(w2t + (size_t)col * 384);
    __builtin_prefetch((const void*)(w2t + ((size_t)col + (size_t)stride * 16) * 384), 0, 3);
    v8f acc = {};
    // double-buffered B: issue ks+1's b128 pair before consuming ks's
    v8h lo = wp[bb], hi = wp[bb + 1];
#pragma unroll
    for (int ks = 0; ks < 12; ++ks) {
      v8h nlo = lo, nhi = hi;
      if (ks < 11) { nlo = wp[(ks + 1) * 4 + bb]; nhi = wp[(ks + 1) * 4 + bb + 1]; }
      v16h bfrag;
#pragma unroll
      for (int j = 0; j < 8; ++j) { bfrag[j] = lo[j]; bfrag[8 + j] = hi[j]; }
      acc = __builtin_amdgcn_wmma_f32_16x16x32_f16(false, afrag[ks], false, bfrag,
                                                   (short)0, acc, false, false);
      lo = nlo; hi = nhi;
    }
    tp_consume(S, acc, b2[col], col, lane, layer);
  }
  tp_scatter(S, esrc, node_accum, eb, out_dim);
}

// ------ fallback: same kernel but gathering B straight from f32 w2 ---------
__global__ __launch_bounds__(256)
void fc2_tp_kernel(const _Float16* __restrict__ h, const float* __restrict__ w2,
                   const float* __restrict__ b2, const float* __restrict__ na,
                   const float* __restrict__ sh,
                   const int* esrc, const int* edst,
                   float* node_accum, int layer, int numel, int in_dim, int out_dim) {
  __shared__ TpShared S;
  int eb = blockIdx.y;
  tp_stage(S, h, na, sh, edst, eb, in_dim);

  int tid = threadIdx.x;
  int wid = tid >> 5, lane = tid & 31;
  v16h afrag[12];
#pragma unroll
  for (int ks = 0; ks < 12; ++ks)
    afrag[ks] = frag_a_lds(&S.h[0][ks * 32], 384, lane);

  int nchunks = numel >> 4;
  int stride  = gridDim.x * 8;
  int n   = lane & 15;
  int kb2 = (lane >> 4) << 4;

  for (int ch = blockIdx.x * 8 + wid; ch < nchunks; ch += stride) {
    int col = ch * 16 + n;
    v8f acc = {};
#pragma unroll
    for (int ks = 0; ks < 12; ++ks) {
      v16h bfrag;
#pragma unroll
      for (int j = 0; j < 16; ++j)
        bfrag[j] = (_Float16)w2[(size_t)(ks * 32 + kb2 + j) * numel + col];
      acc = __builtin_amdgcn_wmma_f32_16x16x32_f16(false, afrag[ks], false, bfrag,
                                                   (short)0, acc, false, false);
    }
    tp_consume(S, acc, b2[col], col, lane, layer);
  }
  tp_scatter(S, esrc, node_accum, eb, out_dim);
}

// ---------------- BatchNorm stats & apply (+ residual) ---------------------
__global__ __launch_bounds__(256)
void bn_stats_kernel(const float* accum, const float* cnt,
                     float* mean_o, float* scale_o, int layer) {
  __shared__ float red[256];
  int chn = blockIdx.x, tid = threadIdx.x;
  int io = c_ioff[layer], ic = c_icnt[layer];
  int off = 0, d = 1, sub = 0, u = 0, rem = chn;
  for (int i = 0; i < ic; ++i) {
    if (rem < c_irr[io + i].m) {
      off = c_irr[io + i].off; d = c_irr[io + i].d; sub = c_irr[io + i].sub;
      u = rem; break;
    }
    rem -= c_irr[io + i].m;
  }
  float mean = 0.0f;
  if (sub) {
    float s = 0.0f;
    for (int nn = tid; nn < N_NODES; nn += 256)
      s += accum[(size_t)nn * MAXD + off + u] / fmaxf(cnt[nn], 1.0f);
    red[tid] = s; __syncthreads();
    for (int st = 128; st > 0; st >>= 1) { if (tid < st) red[tid] += red[tid + st]; __syncthreads(); }
    mean = red[0] / (float)N_NODES;
    __syncthreads();
  }
  float s2 = 0.0f;
  for (int i = tid; i < N_NODES * d; i += 256) {
    int nn = i / d, k = i % d;
    float f = accum[(size_t)nn * MAXD + off + u * d + k] / fmaxf(cnt[nn], 1.0f) - mean;
    s2 += f * f;
  }
  red[tid] = s2; __syncthreads();
  for (int st = 128; st > 0; st >>= 1) { if (tid < st) red[tid] += red[tid + st]; __syncthreads(); }
  if (tid == 0) {
    float n2 = red[0] / (float)(N_NODES * d);
    float sc = rsqrtf(n2 + 1e-5f);
    for (int k = 0; k < d; ++k) {
      mean_o[off + u * d + k]  = mean;
      scale_o[off + u * d + k] = sc;
    }
  }
}

__global__ void bn_apply_kernel(const float* accum, const float* cnt,
                                const float* mean, const float* scale,
                                float* na, int out_dim) {
  int i = blockIdx.x * blockDim.x + threadIdx.x;
  if (i >= N_NODES * out_dim) return;
  int nn = i / out_dim, d2 = i % out_dim;
  float upd = accum[(size_t)nn * MAXD + d2] / fmaxf(cnt[nn], 1.0f);
  na[(size_t)nn * MAXD + d2] += (upd - mean[d2]) * scale[d2];
}

// ---------------------------------------------------------------------------
extern "C" void kernel_launch(void* const* d_in, const int* in_sizes, int n_in,
                              void* d_out, int out_size, void* d_ws, size_t ws_size,
                              hipStream_t stream) {
  const float* node_f = (const float*)d_in[0];
  const float* edge_f = (const float*)d_in[1];
  const float* pos    = (const float*)d_in[2];
  const float* nuv    = (const float*)d_in[3];
  const int*   esrc   = (const int*)d_in[4];
  const int*   edst   = (const int*)d_in[5];
  const float* emb_w1 = (const float*)d_in[6];
  const float* emb_b1 = (const float*)d_in[7];
  const float* emb_w2 = (const float*)d_in[8];
  const float* emb_b2 = (const float*)d_in[9];
  const float* fw1[3] = {(const float*)d_in[10], (const float*)d_in[14], (const float*)d_in[18]};
  const float* fb1[3] = {(const float*)d_in[11], (const float*)d_in[15], (const float*)d_in[19]};
  const float* fw2[3] = {(const float*)d_in[12], (const float*)d_in[16], (const float*)d_in[20]};
  const float* fb2[3] = {(const float*)d_in[13], (const float*)d_in[17], (const float*)d_in[21]};
  const float* now = (const float*)d_in[22];
  const float* nob = (const float*)d_in[23];
  const float* eow = (const float*)d_in[24];
  const float* eob = (const float*)d_in[25];
  float* out = (float*)d_out;

  // workspace layout (float offsets)
  float* F      = (float*)d_ws;
  float* sh_buf = F;                 // 2048*9     = 18432
  float* eattr  = F + 18432;         // 2048*128   = 262144
  float* eaug   = F + 280576;        // 2048*384   = 786432
  float* na     = F + 1067008;       // 512*448    = 229376
  float* acc    = F + 1296384;       // 512*448    = 229376
  float* cnt    = F + 1525760;       // 512
  float* bnm    = F + 1526272;       // 448
  float* bns    = F + 1526720;       // 448
  _Float16* hbuf = (_Float16*)(F + 1527168);  // 2048*384 halves
  size_t base_bytes = (size_t)1527168 * 4 + (size_t)2048 * 384 * 2;
  _Float16* w2t = (_Float16*)((char*)d_ws + base_bytes);
  size_t need_t = base_bytes + (size_t)48128 * 384 * 2;
  bool use_t = (ws_size >= need_t);

  const int numel[3]  = {20480, 48128, 40960};
  const int indim[3]  = {128, 352, 448};
  const int outdim[3] = {352, 448, 256};
  const int nch[3]    = {288, 320, 256};

  // zero na + acc + cnt (contiguous region)
  int zlen = 229376 + 229376 + 512;
  fill_kernel<<<(zlen + 255) / 256, 256, 0, stream>>>(na, zlen, 0.0f);
  copy_nodef_kernel<<<(N_NODES * NS + 255) / 256, 256, 0, stream>>>(node_f, na);
  count_kernel<<<(N_EDGES + 255) / 256, 256, 0, stream>>>(esrc, cnt);
  edge_init_kernel<<<N_EDGES, 128, 0, stream>>>(edge_f, pos, nuv, esrc, edst,
                                                emb_w1, emb_b1, emb_w2, emb_b2,
                                                sh_buf, eattr);
  for (int L = 0; L < 3; ++L) {
    build_eaug_kernel<<<N_EDGES, 128, 0, stream>>>(eattr, na, esrc, edst, eaug);
    // h = relu(eaug @ w1 + b1), stored f16
    gemm_kernel<true, true><<<dim3(12, 64), 128, 0, stream>>>(
        eaug, 384, fw1[L], fb1[L], (void*)hbuf, N_EDGES, 384, 384);
    fill_kernel<<<(229376 + 255) / 256, 256, 0, stream>>>(acc, 229376, 0.0f);
    if (use_t) {
      transpose_w2_kernel<<<dim3(numel[L] / 32, 12), 256, 0, stream>>>(
          fw2[L], w2t, numel[L]);
      fc2_tp_t_kernel<<<dim3(CSPLIT, N_EDGES / 16), 256, 0, stream>>>(
          hbuf, w2t, fb2[L], na, sh_buf, esrc, edst, acc,
          L, numel[L], indim[L], outdim[L]);
    } else {
      fc2_tp_kernel<<<dim3(CSPLIT, N_EDGES / 16), 256, 0, stream>>>(
          hbuf, fw2[L], fb2[L], na, sh_buf, esrc, edst, acc,
          L, numel[L], indim[L], outdim[L]);
    }
    bn_stats_kernel<<<nch[L], 256, 0, stream>>>(acc, cnt, bnm, bns, L);
    bn_apply_kernel<<<(N_NODES * outdim[L] + 255) / 256, 256, 0, stream>>>(
        acc, cnt, bnm, bns, na, outdim[L]);
  }
  // node_out = na[:, :256] @ now + nob   -> d_out[0 : 512*128]
  gemm_kernel<false, false><<<dim3(4, 16), 128, 0, stream>>>(
      na, MAXD, now, nob, (void*)out, N_NODES, 128, 256);
  // edge_out = eaug @ eow + eob          -> d_out[512*128 : ]
  gemm_kernel<false, false><<<dim3(4, 64), 128, 0, stream>>>(
      eaug, 384, eow, eob, (void*)(out + N_NODES * NS), N_EDGES, 128, 384);
}
